// ComposableMoE_90735479095893
// MI455X (gfx1250) — matrix operand order repaired and checked by reading, not compiled, not run on hardware
//
#include <hip/hip_runtime.h>
#include <hip/hip_bf16.h>

// ---------------- problem constants ----------------
#define NTOK   16384
#define DIN    1024
#define DEMB   128
#define NE     16
#define TOPK   2
#define H1U    1024
#define H2U    512
#define DOUT   512
#define TM     32            // tokens per MLP block (2 row tiles)

// ---------------- vector types ----------------
typedef __attribute__((ext_vector_type(16))) __bf16 v16bf;
typedef __attribute__((ext_vector_type(8)))  float  v8f;
typedef __attribute__((ext_vector_type(8)))  short  v8s;
typedef __attribute__((ext_vector_type(16))) short  v16s;

// fp32 -> bf16 round-to-nearest-even (bits)
__device__ __forceinline__ unsigned short f2bf(float f) {
    unsigned int u = __builtin_bit_cast(unsigned int, f);
    u += 0x7fffu + ((u >> 16) & 1u);
    return (unsigned short)(u >> 16);
}

// A fragment: 16x32 bf16, row-major source with leading dim ld (elements).
// ISA layout: lane<16 holds row=lane, K = {kk+0..7, kk+16..23};
//             lane>=16 holds row=lane-16, K = {kk+8..15, kk+24..31}.
__device__ __forceinline__ v16bf load_frag_a(const unsigned short* base, int ld,
                                             int kk, int lane) {
    int r  = lane & 15;
    int kh = (lane < 16) ? 0 : 8;
    const unsigned short* p0 = base + (long)r * ld + kk + kh;
    v8s lo = *(const v8s*)p0;
    v8s hi = *(const v8s*)(p0 + 16);
    v16s f = __builtin_shufflevector(lo, hi, 0,1,2,3,4,5,6,7,8,9,10,11,12,13,14,15);
    return __builtin_bit_cast(v16bf, f);
}

// B fragment: 32x16 bf16 where memory is transposed weights [n][k] (ld over k).
// ISA layout: col = lane&15, K = {kk+0..7, kk+16..23} (lane<16) / {kk+8..15, kk+24..31} (lane>=16).
__device__ __forceinline__ v16bf load_frag_b(const unsigned short* base, int ld,
                                             int kk, int lane) {
    int n  = lane & 15;
    int kh = (lane < 16) ? 0 : 8;
    const unsigned short* p0 = base + (long)n * ld + kk + kh;
    v8s lo = *(const v8s*)p0;
    v8s hi = *(const v8s*)(p0 + 16);
    v16s f = __builtin_shufflevector(lo, hi, 0,1,2,3,4,5,6,7,8,9,10,11,12,13,14,15);
    return __builtin_bit_cast(v16bf, f);
}

__device__ __forceinline__ v8f wmma_bf16(v16bf a, v16bf b, v8f c) {
    return __builtin_amdgcn_wmma_f32_16x16x32_bf16(false, a, false, b,
                                                   (short)0, c, false, false);
}

// relu + bf16 store of one 16x16 C tile; dst points at row 0 of the tile's rows
__device__ __forceinline__ void store_tile_relu(unsigned short* dst, int ldd,
                                                v8f acc, int n, int mbase, float bias) {
    #pragma unroll
    for (int r = 0; r < 8; ++r) {
        float v = acc[r] + bias;
        dst[(long)(mbase + r) * ldd + n] = f2bf(v > 0.f ? v : 0.f);
    }
}

// ---------------- kernel: fp32 -> bf16 straight convert ----------------
__global__ __launch_bounds__(256) void k_cvt(const float* __restrict__ in,
                                             unsigned short* __restrict__ out,
                                             long n) {
    long i = (long)blockIdx.x * blockDim.x + threadIdx.x;
    if (i < n) out[i] = f2bf(in[i]);
}

// ---------------- kernel: tiled transpose + convert ----------------
// in slice:  R x C fp32 row-major ; out slice: C x R bf16 row-major
__global__ __launch_bounds__(256) void k_transpose_cvt(const float* __restrict__ in,
                                                       unsigned short* __restrict__ out,
                                                       int R, int C) {
    __shared__ unsigned short t[32][33];
    const float* ip = in + (long)blockIdx.z * R * C;
    unsigned short* op = out + (long)blockIdx.z * R * C;
    int c0 = blockIdx.x * 32, r0 = blockIdx.y * 32;
    int tx = threadIdx.x & 31, ty = threadIdx.x >> 5;   // 32 x 8
    #pragma unroll
    for (int i = 0; i < 32; i += 8) {
        int r = r0 + ty + i, c = c0 + tx;
        t[ty + i][tx] = (r < R && c < C) ? f2bf(ip[(long)r * C + c]) : (unsigned short)0;
    }
    __syncthreads();
    #pragma unroll
    for (int i = 0; i < 32; i += 8) {
        int oc = c0 + ty + i, orr = r0 + tx;
        if (oc < C && orr < R) op[(long)oc * R + orr] = t[tx][ty + i];
    }
}

// ---------------- kernel: router GEMM  q = x @ Wr + br ----------------
// 64 tokens per block; wave w owns column tile w (8 x 16 = 128 cols), 4
// row-tile accumulators (B reused 4x). Fully-unrolled ping-pong K loop:
// preloads stay ahead of the consuming WMMAs, no loop-carried registers.
__global__ __launch_bounds__(256) void k_router_gemm(const unsigned short* __restrict__ x_bf,
                                                     const unsigned short* __restrict__ Wrt, // [DEMB][DIN]
                                                     const float* __restrict__ br,
                                                     float* __restrict__ q) {
    int lane = threadIdx.x & 31, wave = threadIdx.x >> 5;
    const unsigned short* arow = x_bf + (long)blockIdx.x * 64 * DIN;
    const unsigned short* brow = Wrt + (long)wave * 16 * DIN;
    v8f acc[4] = {v8f{}, v8f{}, v8f{}, v8f{}};

    v16bf pb = load_frag_b(brow, DIN, 0, lane);
    v16bf pa[4];
    #pragma unroll
    for (int rt = 0; rt < 4; ++rt)
        pa[rt] = load_frag_a(arow + (long)rt * 16 * DIN, DIN, 0, lane);

    #pragma unroll
    for (int kk = 0; kk < DIN; kk += 64) {
        int k1 = kk + 32;                              // always < DIN
        v16bf qb = load_frag_b(brow, DIN, k1, lane);
        v16bf qa[4];
        #pragma unroll
        for (int rt = 0; rt < 4; ++rt)
            qa[rt] = load_frag_a(arow + (long)rt * 16 * DIN, DIN, k1, lane);
        #pragma unroll
        for (int rt = 0; rt < 4; ++rt)
            acc[rt] = wmma_bf16(pa[rt], pb, acc[rt]);

        int k2 = (kk + 64 < DIN) ? kk + 64 : kk;       // folds away under unroll
        pb = load_frag_b(brow, DIN, k2, lane);
        #pragma unroll
        for (int rt = 0; rt < 4; ++rt)
            pa[rt] = load_frag_a(arow + (long)rt * 16 * DIN, DIN, k2, lane);
        #pragma unroll
        for (int rt = 0; rt < 4; ++rt)
            acc[rt] = wmma_bf16(qa[rt], qb, acc[rt]);
    }
    int n = wave * 16 + (lane & 15);
    float bias = br[n];
    int mbase = (lane < 16) ? 0 : 8;
    #pragma unroll
    for (int rt = 0; rt < 4; ++rt)
        #pragma unroll
        for (int r = 0; r < 8; ++r)
            q[(long)(blockIdx.x * 64 + rt * 16 + mbase + r) * DEMB + n] = acc[rt][r] + bias;
}

// ---------------- kernel: distances, top-2, gates, expert lists ----------------
// One wave per token (8 tokens / block). Lane l owns dims 4l..4l+3.
__global__ __launch_bounds__(256) void k_router_topk(const float* __restrict__ q,
                                                     const float* __restrict__ emb, // [NE][DEMB]
                                                     int* __restrict__ counts,
                                                     int* __restrict__ elist,      // [NE][NTOK]
                                                     int* __restrict__ tok_e,      // [NTOK][2]
                                                     int* __restrict__ tok_slot,   // [NTOK][2]
                                                     float* __restrict__ tok_gate) // [NTOK][2]
{
    __shared__ float es[NE][DEMB];
    for (int i = threadIdx.x; i < NE * DEMB; i += 256) es[i / DEMB][i % DEMB] = emb[i];
    __syncthreads();
    int lane = threadIdx.x & 31, wave = threadIdx.x >> 5;
    int n = blockIdx.x * 8 + wave;
    float qa[4];
    #pragma unroll
    for (int j = 0; j < 4; ++j) qa[j] = q[(long)n * DEMB + lane * 4 + j];
    float d2[NE];
    #pragma unroll
    for (int e = 0; e < NE; ++e) {
        float s = 0.f;
        #pragma unroll
        for (int j = 0; j < 4; ++j) { float d = qa[j] - es[e][lane * 4 + j]; s += d * d; }
        d2[e] = s;
    }
    for (int off = 16; off >= 1; off >>= 1)
        #pragma unroll
        for (int e = 0; e < NE; ++e) d2[e] += __shfl_down(d2[e], off, 32);
    if (lane == 0) {
        int i0 = 0; float s0 = -d2[0];
        #pragma unroll
        for (int e = 1; e < NE; ++e) { float s = -d2[e]; if (s > s0) { s0 = s; i0 = e; } }
        int i1 = (i0 == 0) ? 1 : 0; float s1 = -d2[i1];
        #pragma unroll
        for (int e = 0; e < NE; ++e) {
            if (e == i0) continue;
            float s = -d2[e]; if (s > s1) { s1 = s; i1 = e; }
        }
        float e1 = __expf(s1 - s0);          // s0 >= s1
        float inv = 1.0f / (1.0f + e1);
        int p0 = atomicAdd(&counts[i0], 1);
        int p1 = atomicAdd(&counts[i1], 1);
        elist[i0 * NTOK + p0] = n;
        elist[i1 * NTOK + p1] = n;
        tok_e[n * 2 + 0] = i0;  tok_e[n * 2 + 1] = i1;
        tok_slot[n * 2 + 0] = p0; tok_slot[n * 2 + 1] = p1;
        tok_gate[n * 2 + 0] = inv; tok_gate[n * 2 + 1] = e1 * inv;
    }
}

// ---------------- kernel: 16-entry exclusive scan ----------------
__global__ void k_scan(const int* __restrict__ counts, int* __restrict__ bases) {
    if (threadIdx.x == 0) {
        int s = 0;
        for (int e = 0; e < NE; ++e) { bases[e] = s; s += counts[e]; }
        bases[NE] = s;
    }
}

// 2x2-blocked, fully-unrolled ping-pong GEMM tile accumulation.
// A from As0/As1 (16-row tiles, LDS), B from Bp0/Bp1 (transposed weights).
// K must be a multiple of 64.
#define GEMM_2X2_PIPELINED(As0, As1, Bp0, Bp1, LD, K)                          \
    {                                                                          \
        v16bf pa0 = load_frag_a((As0), (LD), 0, lane);                         \
        v16bf pa1 = load_frag_a((As1), (LD), 0, lane);                         \
        v16bf pb0 = load_frag_b((Bp0), (LD), 0, lane);                         \
        v16bf pb1 = load_frag_b((Bp1), (LD), 0, lane);                         \
        _Pragma("unroll")                                                      \
        for (int kk = 0; kk < (K); kk += 64) {                                 \
            int k1 = kk + 32;                                                  \
            v16bf qa0 = load_frag_a((As0), (LD), k1, lane);                    \
            v16bf qa1 = load_frag_a((As1), (LD), k1, lane);                    \
            v16bf qb0 = load_frag_b((Bp0), (LD), k1, lane);                    \
            v16bf qb1 = load_frag_b((Bp1), (LD), k1, lane);                    \
            a00 = wmma_bf16(pa0, pb0, a00);                                    \
            a10 = wmma_bf16(pa1, pb0, a10);                                    \
            a01 = wmma_bf16(pa0, pb1, a01);                                    \
            a11 = wmma_bf16(pa1, pb1, a11);                                    \
            int k2 = (kk + 64 < (K)) ? kk + 64 : kk;                           \
            pa0 = load_frag_a((As0), (LD), k2, lane);                          \
            pa1 = load_frag_a((As1), (LD), k2, lane);                          \
            pb0 = load_frag_b((Bp0), (LD), k2, lane);                          \
            pb1 = load_frag_b((Bp1), (LD), k2, lane);                          \
            a00 = wmma_bf16(qa0, qb0, a00);                                    \
            a10 = wmma_bf16(qa1, qb0, a10);                                    \
            a01 = wmma_bf16(qa0, qb1, a01);                                    \
            a11 = wmma_bf16(qa1, qb1, a11);                                    \
        }                                                                      \
    }

// ---------------- kernel: sparse 3-layer expert MLP ----------------
// Block = (expert, 32-token tile). 256 threads = 8 waves. 2x2 register
// blocking per wave (2 loads/WMMA) + fully-unrolled ping-pong K loops.
// h2 aliases the dead x buffer to stay at 128KB LDS.
__global__ __launch_bounds__(256) void k_moe_mlp(const unsigned short* __restrict__ x_bf,  // [NTOK][DIN]
                                                 const unsigned short* __restrict__ W1t,   // [NE][H1U][DIN]
                                                 const float* __restrict__ b1,             // [NE][H1U]
                                                 const unsigned short* __restrict__ W2t,   // [NE][H2U][H1U]
                                                 const float* __restrict__ b2,             // [NE][H2U]
                                                 const unsigned short* __restrict__ W3t,   // [NE][DOUT][H2U]
                                                 const float* __restrict__ b3,             // [NE][DOUT]
                                                 const int* __restrict__ counts,
                                                 const int* __restrict__ bases,
                                                 const int* __restrict__ elist,
                                                 float* __restrict__ eo)                   // [NTOK*TOPK][DOUT]
{
    __shared__ __align__(16) unsigned short smem[TM * DIN + TM * H1U];  // 128 KB
    unsigned short* xs  = smem;                 // [TM][DIN]  (64 KB)
    unsigned short* h1s = smem + TM * DIN;      // [TM][H1U]  (64 KB)
    unsigned short* h2s = smem;                 // [TM][H2U]  (32 KB, aliases xs)
    __shared__ int toks[TM];

    int e = blockIdx.y;
    int tile = blockIdx.x;
    int cnt = counts[e];
    if (tile * TM >= cnt) return;
    int rows = cnt - tile * TM; if (rows > TM) rows = TM;

    int tid = threadIdx.x, lane = tid & 31, wave = tid >> 5;
    int mbase = (lane < 16) ? 0 : 8;
    int nlo = lane & 15;

    if (tid < TM) {
        int r = (tid < rows) ? tid : rows - 1;   // clamp padding rows
        toks[tid] = elist[e * NTOK + tile * TM + r];
    }
    __syncthreads();

    // gather token rows of x (bf16) into LDS, 16B per transaction
    for (int i = tid; i < TM * (DIN / 8); i += 256) {
        int r = i / (DIN / 8), c8 = i % (DIN / 8);
        *(v8s*)&xs[(long)r * DIN + c8 * 8] = *(const v8s*)&x_bf[(long)toks[r] * DIN + c8 * 8];
    }
    __syncthreads();

    // ---- layer 1: h1 = relu(x @ W1 + b1)  [TM x 1024] ----
    const unsigned short* w1e = W1t + (long)e * H1U * DIN;
    for (int cp = 0; cp < 4; ++cp) {
        int ct0 = wave * 8 + cp * 2;
        const unsigned short* bp0 = w1e + (long)ct0 * 16 * DIN;
        const unsigned short* bp1 = bp0 + (long)16 * DIN;
        v8f a00 = {}, a01 = {}, a10 = {}, a11 = {};
        GEMM_2X2_PIPELINED(xs, xs + 16 * DIN, bp0, bp1, DIN, DIN);
        int n0 = ct0 * 16 + nlo, n1 = n0 + 16;
        float bi0 = b1[e * H1U + n0], bi1 = b1[e * H1U + n1];
        store_tile_relu(h1s,            H1U, a00, n0, mbase, bi0);
        store_tile_relu(h1s + 16 * H1U, H1U, a10, n0, mbase, bi0);
        store_tile_relu(h1s,            H1U, a01, n1, mbase, bi1);
        store_tile_relu(h1s + 16 * H1U, H1U, a11, n1, mbase, bi1);
    }
    __syncthreads();

    // ---- layer 2: h2 = relu(h1 @ W2 + b2)  [TM x 512] (writes over xs) ----
    const unsigned short* w2e = W2t + (long)e * H2U * H1U;
    for (int cp = 0; cp < 2; ++cp) {
        int ct0 = wave * 4 + cp * 2;
        const unsigned short* bp0 = w2e + (long)ct0 * 16 * H1U;
        const unsigned short* bp1 = bp0 + (long)16 * H1U;
        v8f a00 = {}, a01 = {}, a10 = {}, a11 = {};
        GEMM_2X2_PIPELINED(h1s, h1s + 16 * H1U, bp0, bp1, H1U, H1U);
        int n0 = ct0 * 16 + nlo, n1 = n0 + 16;
        float bi0 = b2[e * H2U + n0], bi1 = b2[e * H2U + n1];
        store_tile_relu(h2s,            H2U, a00, n0, mbase, bi0);
        store_tile_relu(h2s + 16 * H2U, H2U, a10, n0, mbase, bi0);
        store_tile_relu(h2s,            H2U, a01, n1, mbase, bi1);
        store_tile_relu(h2s + 16 * H2U, H2U, a11, n1, mbase, bi1);
    }
    __syncthreads();

    // ---- layer 3: eo = h2 @ W3 + b3  [TM x 512] (ungated, compacted rows) ----
    const unsigned short* w3e = W3t + (long)e * DOUT * H2U;
    int gbase = bases[e] + tile * TM;
    for (int cp = 0; cp < 2; ++cp) {
        int ct0 = wave * 4 + cp * 2;
        const unsigned short* bp0 = w3e + (long)ct0 * 16 * H2U;
        const unsigned short* bp1 = bp0 + (long)16 * H2U;
        v8f a00 = {}, a01 = {}, a10 = {}, a11 = {};
        GEMM_2X2_PIPELINED(h2s, h2s + 16 * H2U, bp0, bp1, H2U, H2U);
        int n0 = ct0 * 16 + nlo, n1 = n0 + 16;
        float bi0 = b3[e * DOUT + n0], bi1 = b3[e * DOUT + n1];
        #pragma unroll
        for (int r = 0; r < 8; ++r) {
            int m0 = mbase + r, m1 = 16 + mbase + r;
            if (m0 < rows) {
                eo[(long)(gbase + m0) * DOUT + n0] = a00[r] + bi0;
                eo[(long)(gbase + m0) * DOUT + n1] = a01[r] + bi1;
            }
            if (m1 < rows) {
                eo[(long)(gbase + m1) * DOUT + n0] = a10[r] + bi0;
                eo[(long)(gbase + m1) * DOUT + n1] = a11[r] + bi1;
            }
        }
    }
}

// ---------------- kernel: gated combine ----------------
__global__ __launch_bounds__(256) void k_combine(const float* __restrict__ eo,
                                                 const int* __restrict__ tok_e,
                                                 const int* __restrict__ tok_slot,
                                                 const float* __restrict__ tok_gate,
                                                 const int* __restrict__ bases,
                                                 float* __restrict__ out) {
    int n = blockIdx.x;
    long r0 = (long)bases[tok_e[n * 2 + 0]] + tok_slot[n * 2 + 0];
    long r1 = (long)bases[tok_e[n * 2 + 1]] + tok_slot[n * 2 + 1];
    float g0 = tok_gate[n * 2 + 0], g1 = tok_gate[n * 2 + 1];
    for (int c = threadIdx.x; c < DOUT; c += 256)
        out[(long)n * DOUT + c] = g0 * eo[r0 * DOUT + c] + g1 * eo[r1 * DOUT + c];
}

// ---------------- host launcher ----------------
extern "C" void kernel_launch(void* const* d_in, const int* in_sizes, int n_in,
                              void* d_out, int out_size, void* d_ws, size_t ws_size,
                              hipStream_t stream) {
    const float* x   = (const float*)d_in[0];
    const float* Wr  = (const float*)d_in[1];
    const float* br  = (const float*)d_in[2];
    const float* emb = (const float*)d_in[3];
    const float* W1  = (const float*)d_in[4];
    const float* b1  = (const float*)d_in[5];
    const float* W2  = (const float*)d_in[6];
    const float* b2  = (const float*)d_in[7];
    const float* W3  = (const float*)d_in[8];
    const float* b3  = (const float*)d_in[9];
    float* out = (float*)d_out;

    // workspace carve-up (256B aligned)
    char* base = (char*)d_ws;
    size_t off = 0;
    auto take = [&](size_t bytes) {
        char* p = base + off;
        off = (off + bytes + 255) & ~(size_t)255;
        return p;
    };
    unsigned short* x_bf = (unsigned short*)take((size_t)NTOK * DIN * 2);
    unsigned short* Wrt  = (unsigned short*)take((size_t)DEMB * DIN * 2);
    unsigned short* W1t  = (unsigned short*)take((size_t)NE * H1U * DIN * 2);
    unsigned short* W2t  = (unsigned short*)take((size_t)NE * H2U * H1U * 2);
    unsigned short* W3t  = (unsigned short*)take((size_t)NE * DOUT * H2U * 2);
    float* q             = (float*)take((size_t)NTOK * DEMB * 4);
    int*   counts        = (int*)take(NE * 4);
    int*   bases         = (int*)take((NE + 1) * 4);
    int*   elist         = (int*)take((size_t)NE * NTOK * 4);
    int*   tok_e         = (int*)take((size_t)NTOK * 2 * 4);
    int*   tok_slot      = (int*)take((size_t)NTOK * 2 * 4);
    float* tok_gate      = (float*)take((size_t)NTOK * 2 * 4);
    float* eo            = (float*)take((size_t)NTOK * TOPK * DOUT * 4);
    (void)ws_size; (void)in_sizes; (void)n_in; (void)out_size;

    hipMemsetAsync(counts, 0, NE * sizeof(int), stream);

    // convert x to bf16
    {
        long n = (long)NTOK * DIN;
        k_cvt<<<dim3((unsigned)((n + 255) / 256)), dim3(256), 0, stream>>>(x, x_bf, n);
    }
    // transpose+convert weights: out[n][k] layouts for contiguous-K B fragments
    k_transpose_cvt<<<dim3(DEMB / 32, DIN / 32, 1),  dim3(256), 0, stream>>>(Wr, Wrt, DIN, DEMB);
    k_transpose_cvt<<<dim3(H1U / 32,  DIN / 32, NE), dim3(256), 0, stream>>>(W1, W1t, DIN, H1U);
    k_transpose_cvt<<<dim3(H2U / 32,  H1U / 32, NE), dim3(256), 0, stream>>>(W2, W2t, H1U, H2U);
    k_transpose_cvt<<<dim3(DOUT / 32, H2U / 32, NE), dim3(256), 0, stream>>>(W3, W3t, H2U, DOUT);

    // router
    k_router_gemm<<<dim3(NTOK / 64), dim3(256), 0, stream>>>(x_bf, Wrt, br, q);
    k_router_topk<<<dim3(NTOK / 8), dim3(256), 0, stream>>>(q, emb, counts, elist,
                                                            tok_e, tok_slot, tok_gate);
    k_scan<<<dim3(1), dim3(32), 0, stream>>>(counts, bases);

    // sparse expert MLP (blocks with no assigned tile exit immediately)
    k_moe_mlp<<<dim3(NTOK / TM, NE), dim3(256), 0, stream>>>(x_bf, W1t, b1, W2t, b2,
                                                             W3t, b3, counts, bases, elist, eo);
    // gated combine into d_out
    k_combine<<<dim3(NTOK), dim3(256), 0, stream>>>(eo, tok_e, tok_slot, tok_gate, bases, out);
}
